// AdaptiveFilterAttention_19146964206014
// MI455X (gfx1250) — compile-verified, matrix operand order checked by reading, not run on hardware
//
#include <hip/hip_runtime.h>
#include <hip/hip_bf16.h>

typedef __attribute__((ext_vector_type(16))) _Float16 v16h;
typedef __attribute__((ext_vector_type(8)))  float    v8f;
typedef unsigned int v4u __attribute__((ext_vector_type(4)));
typedef int          v8s __attribute__((ext_vector_type(8)));
typedef int          v4s __attribute__((ext_vector_type(4)));

constexpr int B_  = 2;
constexpr int T_  = 2048;
constexpr int D_  = 1024;
constexpr int H_  = 16;
constexpr int HD_ = 64;
constexpr int BT_ = B_ * T_;

#if __has_builtin(__builtin_amdgcn_tensor_load_to_lds)
#define HAVE_TDM 1
#else
#define HAVE_TDM 0
#endif

// ---------------------------------------------------------------------------
// WMMA fragment loaders (CDNA5 16-bit A/B layout, ISA 7.12.2):
// lane = row (A: M, B: N); lanes 0-15 hold K 0..7 & 16..23 as packed pairs,
// lanes 16-31 hold K 8..15 & 24..31.  k(i,h) = 2i + (i>=4 ? 8:0) + 8h.
// ---------------------------------------------------------------------------
__device__ __forceinline__ v16h frag_rowmajor(const _Float16* __restrict__ src, int stride) {
  int lane = threadIdx.x & 31;
  int m = lane & 15, h = lane >> 4;
  const _Float16* p = src + m * stride;
  v16h r;
#pragma unroll
  for (int i = 0; i < 8; ++i) {
    int k = 2 * i + (i >= 4 ? 8 : 0) + 8 * h;
    r[2 * i]     = p[k];
    r[2 * i + 1] = p[k + 1];
  }
  return r;
}

// B-operand gathered from a row-major [K][stride] tile: lane n = fixed column,
// K walks rows.
__device__ __forceinline__ v16h frag_colmajor(const _Float16* __restrict__ src, int stride) {
  int lane = threadIdx.x & 31;
  int n = lane & 15, h = lane >> 4;
  v16h r;
#pragma unroll
  for (int i = 0; i < 8; ++i) {
    int k = 2 * i + (i >= 4 ? 8 : 0) + 8 * h;
    r[2 * i]     = src[k * stride + n];
    r[2 * i + 1] = src[(k + 1) * stride + n];
  }
  return r;
}

__device__ __forceinline__ v8f wmma_f16(v16h a, v16h b, v8f c) {
  return __builtin_amdgcn_wmma_f32_16x16x32_f16(false, a, false, b, (short)0, c,
                                                false, false);
}

#if HAVE_TDM
// ---------------------------------------------------------------------------
// Tensor Data Mover: DMA a contiguous 2D f16 tile (rows x rowlen, row stride
// gstride elements) from global memory into LDS.  D# packing per ISA 8.3/8.4:
// Group0: count=1, lds_addr, global_addr[56:0], type=2.
// Group1: data_size=1 (2B), tensor_dim0/1, tile_dim0/1, tensor_dim0_stride.
// Wave-level op; TENSORcnt tracked, waited with s_wait_tensorcnt.
// ---------------------------------------------------------------------------
__device__ __forceinline__ void tdm_load_tile(const _Float16* gsrc, _Float16* ldst,
                                              int rows, int rowlen, int gstride) {
  unsigned long long ga = (unsigned long long)(uintptr_t)gsrc;
  unsigned int lds_off  = (unsigned int)(uintptr_t)ldst;  // low 32 bits = LDS offset
  v4u g0;
  g0[0] = 1u;                                             // count=1 valid
  g0[1] = lds_off;
  g0[2] = (unsigned int)(ga & 0xFFFFFFFFu);
  g0[3] = (unsigned int)((ga >> 32) & 0x01FFFFFFu) | (2u << 30);  // type=2 "image"
  v8s g1 = {};
  g1[0] = (int)(1u << 16);                                // data_size = 2 bytes
  g1[1] = (int)((unsigned)(rowlen & 0xFFFF) << 16);       // tensor_dim0[15:0]
  g1[2] = (int)((unsigned)(T_ & 0xFFFF) << 16);           // tensor_dim1[15:0]
  g1[3] = (int)((unsigned)(rowlen & 0xFFFF) << 16);       // tile_dim0
  g1[4] = rows;                                           // tile_dim1
  g1[5] = gstride;                                        // tensor_dim0_stride[31:0]
  v4s z4 = {};
#if __clang_major__ >= 23
  v8s z8 = {};
  __builtin_amdgcn_tensor_load_to_lds(g0, g1, z4, z4, z8, 0);
#else
  __builtin_amdgcn_tensor_load_to_lds(g0, g1, z4, z4, 0);
#endif
}
#endif

// ---------------------------------------------------------------------------
// fp32 -> f16 conversion
// ---------------------------------------------------------------------------
__global__ __launch_bounds__(256) void cvt_f32_f16(const float* __restrict__ src,
                                                   _Float16* __restrict__ dst, int n) {
  int i      = blockIdx.x * blockDim.x + threadIdx.x;
  int stride = gridDim.x * blockDim.x;
  for (; i < n; i += stride) dst[i] = (_Float16)src[i];
}

// ---------------------------------------------------------------------------
// QKV projection GEMM: P[M,N] = X[M,K] * W[N,K]^T + bias, f16 out in
// [B,H,T,HD] layout.  One wave per 16x64 strip: A fragment reused across 4
// B fragments -> 4 WMMAs per 32-deep K step (A traffic cut 4x).
// ---------------------------------------------------------------------------
__global__ __launch_bounds__(256) void gemm_qkv(const _Float16* __restrict__ Xh,
                                                const _Float16* __restrict__ Wh,
                                                const float* __restrict__ bias,
                                                _Float16* __restrict__ Out) {
  const int wave  = threadIdx.x >> 5;
  const int strip = blockIdx.x * 8 + wave;          // 4096 strips
  const int mt    = strip / (D_ / 64);
  const int nt4   = strip % (D_ / 64);
  const _Float16* arow = Xh + (size_t)mt * 16 * D_;
  const _Float16* brow = Wh + (size_t)nt4 * 64 * D_;

  v8f c[4] = {};
#pragma unroll 2
  for (int k0 = 0; k0 < D_; k0 += 32) {
    __builtin_prefetch(arow + k0 + 128, 0, 1);
    v16h a = frag_rowmajor(arow + k0, D_);
#pragma unroll
    for (int t = 0; t < 4; ++t) {
      v16h b = frag_rowmajor(brow + (size_t)t * 16 * D_ + k0, D_);
      c[t] = wmma_f16(a, b, c[t]);
    }
  }

  const int lane = threadIdx.x & 31;
  const int n15  = lane & 15, hh = lane >> 4;
#pragma unroll
  for (int t = 0; t < 4; ++t) {
    const int n    = nt4 * 64 + t * 16 + n15;
    const float bv = bias[n];
    const int hsel = n >> 6;   // / HD
    const int hd   = n & 63;   // % HD
#pragma unroll
    for (int j = 0; j < 8; ++j) {
      int row = mt * 16 + j + 8 * hh;          // row in [0, B*T)
      int b_  = row >> 11;                     // / T
      int tt  = row & (T_ - 1);                // % T
      Out[(((size_t)b_ * H_ + hsel) * T_ + tt) * HD_ + hd] = (_Float16)(c[t][j] + bv);
    }
  }
}

// ---------------------------------------------------------------------------
// Flash attention with analytic |lag| decay.  Block = 4 waves, each wave owns
// one 16-query strip; K/V tiles (32 keys x 64 hd, f16) brought into LDS by
// the Tensor Data Mover (wave 0 issues, s_wait_tensorcnt, barrier) and shared
// by all 4 waves.  Online softmax in fp32.
// ---------------------------------------------------------------------------
__global__ __launch_bounds__(128) void flash_decay_attn(
    const _Float16* __restrict__ qh, const _Float16* __restrict__ kh,
    const _Float16* __restrict__ vh, const float* __restrict__ alpha_p,
    const float* __restrict__ sigma_p, const float* __restrict__ eta_p,
    _Float16* __restrict__ oh /* [B,T,D] f16 */) {
  __shared__ _Float16 sK[32 * 64];
  __shared__ _Float16 sV[32 * 64];
  __shared__ _Float16 sP[4 * 16 * 32];

  const float alpha  = alpha_p[0];
  const float eta    = eta_p[0];
  const float dscale = 1.0f / (sigma_p[0] + 1e-8f);
  const float rate   = alpha + eta;           // exp(-a*l)*exp(-e*l) fused
  const float sscale = 0.125f;                // HD^-0.5

  const int tid  = threadIdx.x;
  const int wave = tid >> 5;
  const int lane = tid & 31;
  const int n15  = lane & 15, hh = lane >> 4;

  const int nqb  = T_ / 64;
  const int bh   = blockIdx.x / nqb;
  const int qblk = blockIdx.x % nqb;
  const int b    = bh / H_;
  const int h    = bh % H_;

  const _Float16* Qb = qh + (size_t)bh * T_ * HD_;
  const _Float16* Kb = kh + (size_t)bh * T_ * HD_;
  const _Float16* Vb = vh + (size_t)bh * T_ * HD_;

  const int q0 = qblk * 64 + wave * 16;
  const v16h aq0 = frag_rowmajor(Qb + (size_t)q0 * HD_, HD_);       // hd 0..31
  const v16h aq1 = frag_rowmajor(Qb + (size_t)q0 * HD_ + 32, HD_);  // hd 32..63

  float mrow[8], lrow[8];
  v8f acc[4] = {};
#pragma unroll
  for (int j = 0; j < 8; ++j) { mrow[j] = -1e30f; lrow[j] = 0.0f; }

  _Float16* myP = sP + wave * (16 * 32);

  for (int kb = 0; kb < T_ / 32; ++kb) {
    __syncthreads();  // previous iteration fully consumed LDS K/V
#if HAVE_TDM
    if (wave == 0) {
      tdm_load_tile(Kb + (size_t)kb * 32 * HD_, sK, 32, HD_, HD_);
      tdm_load_tile(Vb + (size_t)kb * 32 * HD_, sV, 32, HD_, HD_);
      __builtin_amdgcn_s_wait_tensorcnt(0);
    }
#else
    {  // cooperative 128-thread load: 32x64 halves = 256 uint4 per tile
      const uint4* ks = (const uint4*)(Kb + (size_t)kb * 32 * HD_);
      const uint4* vs = (const uint4*)(Vb + (size_t)kb * 32 * HD_);
      uint4* kd = (uint4*)sK;
      uint4* vd = (uint4*)sV;
      kd[tid] = ks[tid]; kd[tid + 128] = ks[tid + 128];
      vd[tid] = vs[tid]; vd[tid + 128] = vs[tid + 128];
    }
#endif
    __syncthreads();

    // S tiles: keys kb*32 .. kb*32+31
    v8f c0 = {}, c1 = {};
    {
      v16h b00 = frag_rowmajor(sK, 64);             // keys 0-15, hd 0-31
      v16h b01 = frag_rowmajor(sK + 32, 64);        // keys 0-15, hd 32-63
      v16h b10 = frag_rowmajor(sK + 16 * 64, 64);   // keys 16-31
      v16h b11 = frag_rowmajor(sK + 16 * 64 + 32, 64);
      c0 = wmma_f16(aq0, b00, c0);
      c0 = wmma_f16(aq1, b01, c0);
      c1 = wmma_f16(aq0, b10, c1);
      c1 = wmma_f16(aq1, b11, c1);
    }

    const int kg0 = kb * 32 + n15;
    const int kg1 = kg0 + 16;
#pragma unroll
    for (int j = 0; j < 8; ++j) {
      const int qg = q0 + j + 8 * hh;
      float lag0 = fabsf((float)(qg - kg0));
      float lag1 = fabsf((float)(qg - kg1));
      float s0 = c0[j] * sscale * dscale * __expf(-rate * lag0);
      float s1 = c1[j] * sscale * dscale * __expf(-rate * lag1);
      // row max across the 16 lanes of this half-wave
      float vm = fmaxf(s0, s1);
      vm = fmaxf(vm, __shfl_xor(vm, 1, 16));
      vm = fmaxf(vm, __shfl_xor(vm, 2, 16));
      vm = fmaxf(vm, __shfl_xor(vm, 4, 16));
      vm = fmaxf(vm, __shfl_xor(vm, 8, 16));
      float mn = fmaxf(mrow[j], vm);
      float e0 = __expf(s0 - mn);
      float e1 = __expf(s1 - mn);
      float rs = e0 + e1;
      rs += __shfl_xor(rs, 1, 16);
      rs += __shfl_xor(rs, 2, 16);
      rs += __shfl_xor(rs, 4, 16);
      rs += __shfl_xor(rs, 8, 16);
      float co = __expf(mrow[j] - mn);
      lrow[j] = lrow[j] * co + rs;
      mrow[j] = mn;
      acc[0][j] *= co; acc[1][j] *= co; acc[2][j] *= co; acc[3][j] *= co;
      // stage P (16x32 f16, A-operand source) in per-wave LDS region
      const int prow = j + 8 * hh;
      myP[prow * 32 + n15]      = (_Float16)e0;
      myP[prow * 32 + 16 + n15] = (_Float16)e1;
    }

    // same-wave DS RAW: ensure P stores land before fragment gather
    asm volatile("s_wait_dscnt 0" ::: "memory");
    v16h ap = frag_rowmajor(myP, 32);  // P as A operand [16 x 32keys]
#pragma unroll
    for (int blk = 0; blk < 4; ++blk) {
      v16h bv = frag_colmajor(sV + blk * 16, 64);  // V [32keys x 16hd]
      acc[blk] = wmma_f16(ap, bv, acc[blk]);
    }
  }

  // normalize and emit O in [B,T,D] f16 for the output GEMM
#pragma unroll
  for (int j = 0; j < 8; ++j) {
    const float inv = 1.0f / lrow[j];
    const int qg = q0 + j + 8 * hh;
#pragma unroll
    for (int blk = 0; blk < 4; ++blk) {
      const int hd = blk * 16 + n15;
      oh[((size_t)b * T_ + qg) * D_ + h * HD_ + hd] = (_Float16)(acc[blk][j] * inv);
    }
  }
}

// ---------------------------------------------------------------------------
// Output projection: out[M,N] = O[M,K] * Wo[N,K]^T + bo, fp32 output,
// same 16x64-strip structure as gemm_qkv.
// ---------------------------------------------------------------------------
__global__ __launch_bounds__(256) void gemm_out(const _Float16* __restrict__ Oh,
                                                const _Float16* __restrict__ Wh,
                                                const float* __restrict__ bias,
                                                float* __restrict__ out) {
  const int wave  = threadIdx.x >> 5;
  const int strip = blockIdx.x * 8 + wave;
  const int mt    = strip / (D_ / 64);
  const int nt4   = strip % (D_ / 64);
  const _Float16* arow = Oh + (size_t)mt * 16 * D_;
  const _Float16* brow = Wh + (size_t)nt4 * 64 * D_;

  v8f c[4] = {};
#pragma unroll 2
  for (int k0 = 0; k0 < D_; k0 += 32) {
    __builtin_prefetch(arow + k0 + 128, 0, 1);
    v16h a = frag_rowmajor(arow + k0, D_);
#pragma unroll
    for (int t = 0; t < 4; ++t) {
      v16h b = frag_rowmajor(brow + (size_t)t * 16 * D_ + k0, D_);
      c[t] = wmma_f16(a, b, c[t]);
    }
  }

  const int lane = threadIdx.x & 31;
  const int n15  = lane & 15, hh = lane >> 4;
#pragma unroll
  for (int t = 0; t < 4; ++t) {
    const int n    = nt4 * 64 + t * 16 + n15;
    const float bv = bias[n];
#pragma unroll
    for (int j = 0; j < 8; ++j) {
      int row = mt * 16 + j + 8 * hh;
      out[(size_t)row * D_ + n] = c[t][j] + bv;
    }
  }
}

// ---------------------------------------------------------------------------
extern "C" void kernel_launch(void* const* d_in, const int* in_sizes, int n_in,
                              void* d_out, int out_size, void* d_ws, size_t ws_size,
                              hipStream_t stream) {
  const float* x     = (const float*)d_in[0];
  const float* alpha = (const float*)d_in[1];
  const float* sigma = (const float*)d_in[2];
  const float* eta   = (const float*)d_in[3];
  const float* Wq    = (const float*)d_in[4];
  const float* bq    = (const float*)d_in[5];
  const float* Wk    = (const float*)d_in[6];
  const float* bk    = (const float*)d_in[7];
  const float* Wv    = (const float*)d_in[8];
  const float* bv    = (const float*)d_in[9];
  const float* Wo    = (const float*)d_in[10];
  const float* bo    = (const float*)d_in[11];
  float* out = (float*)d_out;

  // workspace layout (f16 elements)
  _Float16* ws  = (_Float16*)d_ws;
  const size_t NXD = (size_t)BT_ * D_;   // 4,194,304
  const size_t NW  = (size_t)D_ * D_;    // 1,048,576
  _Float16* xh  = ws;
  _Float16* wqh = xh + NXD;
  _Float16* wkh = wqh + NW;
  _Float16* wvh = wkh + NW;
  _Float16* woh = wvh + NW;
  _Float16* qhp = woh + NW;
  _Float16* khp = qhp + NXD;
  _Float16* vhp = khp + NXD;
  _Float16* ohp = vhp + NXD;

  cvt_f32_f16<<<2048, 256, 0, stream>>>(x, xh, (int)NXD);
  cvt_f32_f16<<<1024, 256, 0, stream>>>(Wq, wqh, (int)NW);
  cvt_f32_f16<<<1024, 256, 0, stream>>>(Wk, wkh, (int)NW);
  cvt_f32_f16<<<1024, 256, 0, stream>>>(Wv, wvh, (int)NW);
  cvt_f32_f16<<<1024, 256, 0, stream>>>(Wo, woh, (int)NW);

  // 4096 strips (16x64), 8 waves/block
  gemm_qkv<<<512, 256, 0, stream>>>(xh, wqh, bq, qhp);
  gemm_qkv<<<512, 256, 0, stream>>>(xh, wkh, bk, khp);
  gemm_qkv<<<512, 256, 0, stream>>>(xh, wvh, bv, vhp);

  // B*H*(T/64) = 1024 blocks, 4 waves each
  flash_decay_attn<<<1024, 128, 0, stream>>>(qhp, khp, vhp, alpha, sigma, eta, ohp);

  gemm_out<<<512, 256, 0, stream>>>(ohp, woh, bo, out);
}
